// Decoder_30751965839569
// MI455X (gfx1250) — matrix operand order repaired and checked by reading, not compiled, not run on hardware
//
#include <hip/hip_runtime.h>
#include <hip/hip_bf16.h>

typedef __bf16 bf16;
typedef __attribute__((ext_vector_type(16))) __bf16 v16bf;
typedef __attribute__((ext_vector_type(8)))  __bf16 v8bf;
typedef __attribute__((ext_vector_type(8)))  float  v8f;

#define NN 100000
#define CC 256
#define EE 3200000

// ---------------------------------------------------------------- converts
__global__ void k_cvt_bf16(const float* __restrict__ src, bf16* __restrict__ dst, int n) {
    int i = blockIdx.x * blockDim.x + threadIdx.x;
    if (i < n) dst[i] = (bf16)src[i];
}

// deg starts at 1.0 (self loop), pre (fused layer-3 accumulator) starts at 0
__global__ void k_init(float* __restrict__ deg, float* __restrict__ pre, int n) {
    int i = blockIdx.x * blockDim.x + threadIdx.x;
    if (i < n) { deg[i] = 1.0f; pre[i] = 0.0f; }
}

__global__ void k_deg(const long long* __restrict__ col, float* __restrict__ deg, int e) {
    int i = blockIdx.x * blockDim.x + threadIdx.x;
    if (i < e) atomicAdd(&deg[(int)col[i]], 1.0f);
}

__global__ void k_dinv(float* __restrict__ deg, int n) {
    int i = blockIdx.x * blockDim.x + threadIdx.x;
    if (i < n) deg[i] = rsqrtf(deg[i]);   // deg >= 1 always (self loops)
}

// ---------------------------------------------------------------- WMMA core
// One wave computes a 16x16 f32 tile of  A(MxK) @ B(KxN)  where B = W^T,
// i.e. B column n is W row n (contiguous in memory). K = CC = 256.
__device__ __forceinline__ v8f wmma_tile(const bf16* __restrict__ A,
                                         const bf16* __restrict__ W,
                                         int tm, int tn, int lane) {
    v8f acc = {};
    const int half = lane >> 4;       // 0: lanes 0-15, 1: lanes 16-31
    const int lr   = lane & 15;
    const bf16* arow = A + (size_t)(tm * 16 + lr) * CC;   // A row (M = lr)
    const bf16* wrow = W + (size_t)(tn * 16 + lr) * CC;   // B col (N = lr) = W row
#pragma unroll
    for (int k0 = 0; k0 < CC; k0 += 32) {
        // A 16x32 bf16 fragment: elems 0..7 -> K = k0+half*16+0..7
        //                        elems 8..15 -> K = k0+half*16+8..15
        const int ka = k0 + half * 16;
        v8bf a_lo = *(const v8bf*)(arow + ka);
        v8bf a_hi = *(const v8bf*)(arow + ka + 8);
        v16bf a = __builtin_shufflevector(a_lo, a_hi,
                                          0, 1, 2, 3, 4, 5, 6, 7,
                                          8, 9, 10, 11, 12, 13, 14, 15);
        // B 32x16 bf16 fragment: elem i -> K = k0+half*16+i (contiguous W row)
        v16bf b = *(const v16bf*)(wrow + k0 + half * 16);
        acc = __builtin_amdgcn_wmma_f32_16x16x32_bf16(
                  false, a, false, b, (short)0, acc, false, false);
    }
    return acc;
}

// GEMM1: h = x @ W_gcn^T  (no bias, no relu: bias comes after aggregation)
__global__ void k_gemm1(const bf16* __restrict__ xb, const bf16* __restrict__ wb,
                        float* __restrict__ h) {
    const int wave = threadIdx.x >> 5;
    const int lane = threadIdx.x & 31;
    const int tile = blockIdx.x * (blockDim.x >> 5) + wave;
    if (tile >= (NN / 16) * (CC / 16)) return;   // wave-uniform guard
    const int tm = tile >> 4;          // CC/16 == 16 column tiles
    const int tn = tile & 15;
    v8f acc = wmma_tile(xb, wb, tm, tn, lane);
    const int col  = tn * 16 + (lane & 15);
    const int row0 = tm * 16 + ((lane >> 4) << 3);
#pragma unroll
    for (int r = 0; r < 8; ++r)
        h[(size_t)(row0 + r) * CC + col] = acc[r];
}

// GEMM2 fused: t = relu(g @ W1^T + b1); atomically accumulate t*W2 per row
// (fuses layer 3's 256->1 GEMV into the epilogue; saves a 102MB buffer+pass)
__global__ void k_gemm2(const bf16* __restrict__ gb, const bf16* __restrict__ w1b,
                        const float* __restrict__ b1, const float* __restrict__ w2,
                        float* __restrict__ pre) {
    const int wave = threadIdx.x >> 5;
    const int lane = threadIdx.x & 31;
    const int tile = blockIdx.x * (blockDim.x >> 5) + wave;
    if (tile >= (NN / 16) * (CC / 16)) return;
    const int tm = tile >> 4;
    const int tn = tile & 15;
    v8f acc = wmma_tile(gb, w1b, tm, tn, lane);
    const int col  = tn * 16 + (lane & 15);
    const int row0 = tm * 16 + ((lane >> 4) << 3);
    const float bc = b1[col];
    const float wc = w2[col];
#pragma unroll
    for (int r = 0; r < 8; ++r) {
        float v = acc[r] + bc;
        v = v > 0.0f ? v : 0.0f;
        atomicAdd(&pre[row0 + r], v * wc);
    }
}

// ---------------------------------------------------------------- aggregation
// Self-loop term: agg[i][:] = dinv[i]^2 * h[i][:]
__global__ void k_agg_init(const float* __restrict__ h, const float* __restrict__ dinv,
                           float* __restrict__ agg) {
    size_t i = (size_t)blockIdx.x * blockDim.x + threadIdx.x;
    if (i < (size_t)NN * CC) {
        float d = dinv[i >> 8];
        agg[i] = h[i] * d * d;
    }
}

// One wave per edge; lane handles 8 contiguous channels (2x b128 load + 8 f32 atomics)
__global__ void k_edge(const long long* __restrict__ rows, const long long* __restrict__ cols,
                       const float* __restrict__ h, const float* __restrict__ dinv,
                       float* __restrict__ agg, int e) {
    const int wave = threadIdx.x >> 5;
    const int lane = threadIdx.x & 31;
    const int edge = blockIdx.x * (blockDim.x >> 5) + wave;
    if (edge >= e) return;
    const int r = (int)rows[edge];
    const int c = (int)cols[edge];
    const float* hr = h + (size_t)r * CC + lane * 8;
    __builtin_prefetch(hr, 0, 3);                       // global_prefetch_b8
    const float nrm = dinv[r] * dinv[c];
    float4 a0 = *(const float4*)(hr);
    float4 a1 = *(const float4*)(hr + 4);
    float* ob = agg + (size_t)c * CC + lane * 8;
    atomicAdd(ob + 0, a0.x * nrm); atomicAdd(ob + 1, a0.y * nrm);
    atomicAdd(ob + 2, a0.z * nrm); atomicAdd(ob + 3, a0.w * nrm);
    atomicAdd(ob + 4, a1.x * nrm); atomicAdd(ob + 5, a1.y * nrm);
    atomicAdd(ob + 6, a1.z * nrm); atomicAdd(ob + 7, a1.w * nrm);
}

// g = bf16(relu(agg + b_gcn))  -> input of GEMM2
__global__ void k_brc(const float* __restrict__ agg, const float* __restrict__ b,
                      bf16* __restrict__ gb) {
    size_t i = (size_t)blockIdx.x * blockDim.x + threadIdx.x;
    if (i < (size_t)NN * CC) {
        float v = agg[i] + b[i & 255];
        gb[i] = (bf16)(v > 0.0f ? v : 0.0f);
    }
}

// out = sigmoid(relu(pre + b2))
__global__ void k_final(const float* __restrict__ pre, const float* __restrict__ b2,
                        float* __restrict__ out, int n) {
    int i = blockIdx.x * blockDim.x + threadIdx.x;
    if (i < n) {
        float v = pre[i] + b2[0];
        v = v > 0.0f ? v : 0.0f;
        out[i] = 1.0f / (1.0f + __expf(-v));
    }
}

// ---------------------------------------------------------------- launcher
extern "C" void kernel_launch(void* const* d_in, const int* in_sizes, int n_in,
                              void* d_out, int out_size, void* d_ws, size_t ws_size,
                              hipStream_t stream) {
    const float*     x  = (const float*)d_in[0];
    const long long* ei = (const long long*)d_in[1];   // int64 [2, E]
    const float*     Wg = (const float*)d_in[2];
    const float*     bg = (const float*)d_in[3];
    const float*     W1 = (const float*)d_in[4];
    const float*     b1 = (const float*)d_in[5];
    const float*     W2 = (const float*)d_in[6];
    const float*     b2 = (const float*)d_in[7];
    float* out = (float*)d_out;

    char* ws = (char*)d_ws;
    size_t off = 0;
    auto alloc = [&](size_t bytes) -> char* {
        char* p = ws + off;
        off += (bytes + 255) & ~(size_t)255;
        return p;
    };
    bf16*  xb  = (bf16*) alloc((size_t)NN * CC * 2);   // x in bf16
    bf16*  wgb = (bf16*) alloc((size_t)CC * CC * 2);   // W_gcn bf16
    bf16*  w1b = (bf16*) alloc((size_t)CC * CC * 2);   // W1 bf16
    float* h   = (float*)alloc((size_t)NN * CC * 4);   // x @ Wgcn^T
    float* deg = (float*)alloc((size_t)NN * 4);        // deg -> dinv (in place)
    float* pre = (float*)alloc((size_t)NN * 4);        // fused layer-3 accumulator
    float* agg = (float*)alloc((size_t)NN * CC * 4);   // aggregated messages
    bf16*  gb  = (bf16*) alloc((size_t)NN * CC * 2);   // relu(agg+b) bf16

    const long long* rowi = ei;        // sources
    const long long* coli = ei + EE;   // targets

    const int NC = NN * CC;
    k_cvt_bf16<<<(NC + 255) / 256, 256, 0, stream>>>(x,  xb,  NC);
    k_cvt_bf16<<<(CC * CC + 255) / 256, 256, 0, stream>>>(Wg, wgb, CC * CC);
    k_cvt_bf16<<<(CC * CC + 255) / 256, 256, 0, stream>>>(W1, w1b, CC * CC);
    k_init<<<(NN + 255) / 256, 256, 0, stream>>>(deg, pre, NN);
    k_deg<<<(EE + 255) / 256, 256, 0, stream>>>(coli, deg, EE);
    k_dinv<<<(NN + 255) / 256, 256, 0, stream>>>(deg, NN);

    const int tiles  = (NN / 16) * (CC / 16);          // 100000, no tail
    const int gblks  = tiles / 8;                      // 8 waves / block
    k_gemm1<<<gblks, 256, 0, stream>>>(xb, wgb, h);

    k_agg_init<<<NC / 256, 256, 0, stream>>>(h, deg, agg);
    k_edge<<<(EE + 7) / 8, 256, 0, stream>>>(rowi, coli, h, deg, agg, EE);
    k_brc<<<NC / 256, 256, 0, stream>>>(agg, bg, gb);

    k_gemm2<<<gblks, 256, 0, stream>>>(gb, w1b, b1, W2, pre);
    k_final<<<(NN + 255) / 256, 256, 0, stream>>>(pre, b2, out, NN);
}
